// KPClassifier_2516850835757
// MI455X (gfx1250) — matrix operand order repaired and checked by reading, not compiled, not run on hardware
//
#include <hip/hip_runtime.h>

// ---------------- problem constants (from reference setup) ----------------
#define BATCH 2
#define CC    128      // channels in = out
#define NPTS  40960
#define NNB   16       // neighbors
#define KP    15       // kernel points
#define MT    16       // point tile per block
#define XG_STR 136     // padded bf16 row stride for LDS tiles (272B = 17*16B)

typedef __bf16 bf16_t;
typedef bf16_t v16bf __attribute__((ext_vector_type(16)));
typedef bf16_t v8bf  __attribute__((ext_vector_type(8)));
typedef float  v8f   __attribute__((ext_vector_type(8)));
typedef unsigned int u32x4 __attribute__((ext_vector_type(4)));
typedef int          i32x8 __attribute__((ext_vector_type(8)));
typedef int          i32x4 __attribute__((ext_vector_type(4)));

struct U8x32 { uint4 a, b; };   // 32 bytes -> bit_cast to v16bf

// scalar bf16 helper (RNE) for the prep kernels ----------------------------
__device__ __forceinline__ unsigned short f2bf(float f) {
    unsigned u = __float_as_uint(f);
    u += 0x7fffu + ((u >> 16) & 1u);
    return (unsigned short)(u >> 16);
}

// ---------------- prep: x [B][C][N] f32 -> xt [B][N][C] bf16 ---------------
__global__ __launch_bounds__(256) void k_xt(const float* __restrict__ x,
                                            unsigned short* __restrict__ xt) {
    __shared__ float t[32][33];
    const int b = blockIdx.z, c0 = blockIdx.y * 32, n0 = blockIdx.x * 32;
    const int tx = threadIdx.x, ty = threadIdx.y;
    for (int i = ty; i < 32; i += 8)
        t[i][tx] = x[((size_t)b * CC + c0 + i) * NPTS + n0 + tx];
    __syncthreads();
    for (int i = ty; i < 32; i += 8)
        xt[((size_t)b * NPTS + n0 + i) * CC + c0 + tx] = f2bf(t[tx][i]);
}

// ------ prep: weights [K][c][d] f32 -> wbT [K][d][c] bf16; zero shadow row -
__global__ __launch_bounds__(256) void k_wt(const float* __restrict__ w,
                                            unsigned short* __restrict__ wbT,
                                            unsigned short* __restrict__ xt) {
    int idx = blockIdx.x * 256 + threadIdx.x;
    if (blockIdx.x == 0 && threadIdx.x < 64) {   // zero shadow feature row
        ((unsigned*)(xt + (size_t)BATCH * NPTS * CC))[threadIdx.x] = 0u;
    }
    if (idx >= KP * CC * CC) return;
    int k = idx / (CC * CC), r = idx % (CC * CC), d = r / CC, c = r % CC;
    wbT[idx] = f2bf(w[((size_t)k * CC + c) * CC + d]);
}

// ---------------- main fused KPConv (WMMA bf16 + async/TDM) ----------------
__global__ __launch_bounds__(256) void kpconv_main(
    const float* __restrict__ pxyz,          // [B][N][3]
    const int*   __restrict__ pknn,          // [B][N][NNB]
    const float* __restrict__ kpts,          // [KP][3]
    const unsigned short* __restrict__ xt,   // bf16 [B][N][C] (+1 zero row)
    const unsigned short* __restrict__ wbT,  // bf16 [KP][d][c]
    float* __restrict__ tmp)                 // [B][N][C] f32 pre-BN
{
    __shared__ float s_pts[MT][3];
    __shared__ float s_kp[16][3];
    __shared__ int   s_j[MT][NNB];
    __shared__ float s_w[MT][NNB][16];                 // k padded to 16
    __shared__ unsigned short s_xg[MT * NNB][XG_STR];  // gathered bf16 rows
    __shared__ unsigned short s_A[MT][XG_STR];         // A_k tile bf16
    __shared__ unsigned short s_WK[CC][XG_STR];        // W_k^T staged by TDM

    const int tid = threadIdx.x;
    const int b   = blockIdx.y;
    const int m0  = blockIdx.x * MT;

    // --- phase 0: tile coords, kernel points, knn indices ---
    if (tid < MT * 3) { int m = tid / 3, d = tid % 3;
        s_pts[m][d] = pxyz[((size_t)b * NPTS + m0 + m) * 3 + d]; }
    if (tid >= 64 && tid < 64 + KP * 3) { int r = tid - 64;
        s_kp[r / 3][r % 3] = kpts[r]; }
    { int m = tid >> 4, nn = tid & 15;
      s_j[m][nn] = pknn[((size_t)b * NPTS + m0 + m) * NNB + nn]; }
    __syncthreads();

    // --- phase 1: influence weights (one (m,nn) pair per thread) ---
    {
        const int m = tid >> 4, nn = tid & 15;
        const int j = s_j[m][nn];
        const bool valid = (j >= 0 && j < NPTS);
        const int jc = valid ? j : 0;
        const float nx = pxyz[((size_t)b * NPTS + jc) * 3 + 0] - s_pts[m][0];
        const float ny = pxyz[((size_t)b * NPTS + jc) * 3 + 1] - s_pts[m][1];
        const float nz = pxyz[((size_t)b * NPTS + jc) * 3 + 2] - s_pts[m][2];
        #pragma unroll
        for (int k = 0; k < KP; ++k) {
            float dx = nx - s_kp[k][0], dy = ny - s_kp[k][1], dz = nz - s_kp[k][2];
            float d2 = fmaf(dx, dx, fmaf(dy, dy, dz * dz));
            float w  = 1.0f - sqrtf(d2) * (1.0f / 1.2f);
            s_w[m][nn][k] = valid ? fmaxf(w, 0.0f) : 0.0f;
        }
        s_w[m][nn][15] = 0.0f;
    }

    // --- phase 2: gather feature rows with GLOBAL_LOAD_ASYNC_TO_LDS (8B/lane)
    {
        const int wvg = tid >> 5, ln = tid & 31;
        for (int i = 0; i < 32; ++i) {
            const int p = wvg * 32 + i;
            const int j = s_j[p >> 4][p & 15];
            const bool valid = (j >= 0 && j < NPTS);
            const size_t row = valid ? ((size_t)b * NPTS + j)
                                     : (size_t)(BATCH * NPTS);   // zero shadow row
            const unsigned short* g = xt + row * CC + ln * 4;
            unsigned lds_off = (unsigned)(uintptr_t)&s_xg[p][ln * 4];
            asm volatile("global_load_async_to_lds_b64 %0, %1, off"
                         :: "v"(lds_off), "v"(g) : "memory");
        }
        asm volatile("s_wait_asynccnt 0x0" ::: "memory");
    }
    __syncthreads();

    // --- phase 3: per kernel point: TDM-stage W_k, build A_k, WMMA ---
    v8f acc = {};
    const int wv = tid >> 5;            // wave id 0..7 -> output column stripe
    const int ln = tid & 31;
    const int li = ln & 15, lh = ln >> 4;
    const int d0 = wv * 16;

    for (int k = 0; k < KP; ++k) {
        // wave 0 issues one Tensor-Data-Mover copy: W_k (32KB, 8B elems),
        // pad 4 DWORDs per 64 DWORDs -> LDS row stride 272B (conflict-free).
        if (wv == 0) {
            const unsigned long long ga =
                (unsigned long long)(uintptr_t)(wbT + (size_t)k * CC * CC);
            u32x4 g0;
            g0[0] = 1u;                                  // count=1 valid descriptor
            g0[1] = (unsigned)(uintptr_t)&s_WK[0][0];    // lds_addr
            g0[2] = (unsigned)(ga & 0xffffffffu);        // global_addr lo
            g0[3] = (unsigned)(ga >> 32) | (2u << 30);   // global_addr hi | type=2
            i32x8 g1;
            g1[0] = (int)((3u << 16) | (1u << 20) | (5u << 22) | (3u << 25));
            g1[1] = (int)(4096u << 16);                  // tensor_dim0 = 4096 (lo16)
            g1[2] = (int)(1u << 16);                     // tensor_dim1 = 1
            g1[3] = (int)(4096u << 16);                  // tile_dim0 = 4096
            g1[4] = 0;                                   // tile_dim1/2 unused (1D)
            g1[5] = 4096;                                // tensor_dim0_stride
            g1[6] = 0; g1[7] = 0;
            i32x4 gz4 = {0, 0, 0, 0};
            i32x8 gz8 = {0, 0, 0, 0, 0, 0, 0, 0};
            __builtin_amdgcn_tensor_load_to_lds(g0, g1, gz4, gz4, gz8, 0);
        }
        if (k + 1 < KP)
            __builtin_prefetch(wbT + ((size_t)(k + 1) * CC + d0 + li) * CC, 0, 3);

        // 3a: A_k[m,c] = sum_nn w[m,nn,k]*xg[m,nn,c]  (vectorized, overlaps TDM)
        {
            const int m = tid >> 4, c8 = (tid & 15) * 8;
            v8f a8 = {};
            #pragma unroll
            for (int nn = 0; nn < NNB; ++nn) {
                const float w = s_w[m][nn][k];
                const uint4 raw = *(const uint4*)&s_xg[m * NNB + nn][c8];
                const v8bf xb = __builtin_bit_cast(v8bf, raw);
                const v8f  xf = __builtin_convertvector(xb, v8f);
                const v8f  wv8 = {w, w, w, w, w, w, w, w};
                a8 = __builtin_elementwise_fma(xf, wv8, a8);
            }
            const v8bf ab = __builtin_convertvector(a8, v8bf);
            *(uint4*)&s_A[m][c8] = __builtin_bit_cast(uint4, ab);
        }
        if (wv == 0) __builtin_amdgcn_s_wait_tensorcnt(0);
        __syncthreads();

        // 3b: 16x16x128 GEMM slice via 4x v_wmma_f32_16x16x32_bf16 (A,B in LDS)
        {
            #pragma unroll
            for (int q = 0; q < 4; ++q) {
                // A fragment: lane<16 -> K 0-7 & 16-23 of row li; lane>=16 -> 8-15 & 24-31
                U8x32 au;
                au.a = *(const uint4*)&s_A[li][q * 32 + lh * 8];
                au.b = *(const uint4*)&s_A[li][q * 32 + lh * 8 + 16];
                // B fragment: lane<16 -> K 0-7 & 16-23 of column (d0+li); lane>=16 -> 8-15 & 24-31
                const unsigned short* wr = &s_WK[d0 + li][q * 32 + lh * 8];
                U8x32 bu;
                bu.a = *(const uint4*)wr;
                bu.b = *(const uint4*)(wr + 16);
                const v16bf af = __builtin_bit_cast(v16bf, au);
                const v16bf bf = __builtin_bit_cast(v16bf, bu);
                acc = __builtin_amdgcn_wmma_f32_16x16x32_bf16(
                        false, af, false, bf, (short)0, acc, false, false);
            }
        }
        __syncthreads();
    }

    // --- phase 4: store 16x16 f32 tile (C/D layout: VGPR r -> M=r / r+8) ---
    float* orow = tmp + ((size_t)b * NPTS + m0) * CC;
    #pragma unroll
    for (int r = 0; r < 8; ++r)
        orow[(size_t)(r + lh * 8) * CC + d0 + li] = acc[r];
}

// ---------------- BN stats pass 1: deterministic per-block partials --------
__global__ __launch_bounds__(256) void k_stats(const float* __restrict__ tmp,
                                               float* __restrict__ psum,
                                               float* __restrict__ psq) {
    __shared__ float ss[256], sq[256];
    const int bi = blockIdx.x, t = threadIdx.x;
    const int c = t & 127, h = t >> 7;
    const size_t r0 = (size_t)bi * 64 + (size_t)h * 32;
    float s = 0.f, q = 0.f;
    for (int i = 0; i < 32; ++i) {
        float v = tmp[(r0 + i) * CC + c];
        s += v; q = fmaf(v, v, q);
    }
    ss[t] = s; sq[t] = q;
    __syncthreads();
    if (t < 128) {
        psum[(size_t)bi * CC + t] = ss[t] + ss[t + 128];
        psq [(size_t)bi * CC + t] = sq[t] + sq[t + 128];
    }
}

// ---------------- BN stats pass 2: per-channel reduce -> scale/shift -------
__global__ __launch_bounds__(256) void k_finalize(const float* __restrict__ psum,
                                                  const float* __restrict__ psq,
                                                  const float* __restrict__ gamma,
                                                  const float* __restrict__ beta,
                                                  float* __restrict__ scale) {
    __shared__ float ss[256], sq[256];
    const int ch = blockIdx.x, t = threadIdx.x;
    float s = 0.f, q = 0.f;
    for (int i = t; i < (BATCH * NPTS) / 64; i += 256) {
        s += psum[(size_t)i * CC + ch];
        q += psq [(size_t)i * CC + ch];
    }
    ss[t] = s; sq[t] = q; __syncthreads();
    for (int off = 128; off > 0; off >>= 1) {
        if (t < off) { ss[t] += ss[t + off]; sq[t] += sq[t + off]; }
        __syncthreads();
    }
    if (t == 0) {
        const float invn = 1.0f / (float)(BATCH * NPTS);
        const float mean = ss[0] * invn;
        const float var  = sq[0] * invn - mean * mean;
        const float rs   = rsqrtf(var + 1e-5f);
        const float sc   = gamma[ch] * rs;
        scale[ch]      = sc;
        scale[CC + ch] = beta[ch] - mean * sc;
    }
}

// ---------------- apply BN+ReLU, transpose to [B][C][1][N] -----------------
__global__ __launch_bounds__(256) void k_apply(const float* __restrict__ tmp,
                                               const float* __restrict__ scale,
                                               float* __restrict__ out) {
    __shared__ float t[32][33];
    const int b = blockIdx.z, c0 = blockIdx.y * 32, n0 = blockIdx.x * 32;
    const int tx = threadIdx.x, ty = threadIdx.y;
    const int c = c0 + tx;
    const float sc = scale[c], sh = scale[CC + c];
    for (int i = ty; i < 32; i += 8) {
        float v = tmp[((size_t)b * NPTS + n0 + i) * CC + c];
        t[i][tx] = fmaxf(fmaf(v, sc, sh), 0.0f);
    }
    __syncthreads();
    for (int i = ty; i < 32; i += 8)
        out[((size_t)b * CC + c0 + i) * NPTS + n0 + tx] = t[tx][i];
}

// ---------------- workspace layout (bytes) ---------------------------------
static const size_t XT_OFF    = 0;          // 20,971,776  bf16 xt (+zero row)
static const size_t WBT_OFF   = 20971776;   //    491,520  bf16 wbT
static const size_t SCALE_OFF = 21463296;   //      1,024  scale/shift
static const size_t PSUM_OFF  = 21464320;   //    655,360
static const size_t PSQ_OFF   = 22119680;   //    655,360
static const size_t TMP_OFF   = 22775040;   // 41,943,040  f32 pre-BN

extern "C" void kernel_launch(void* const* d_in, const int* in_sizes, int n_in,
                              void* d_out, int out_size, void* d_ws, size_t ws_size,
                              hipStream_t stream) {
    const float* x      = (const float*)d_in[0];  // [B,C,N,1]
    const float* pxyz   = (const float*)d_in[1];  // [B,N,3]
    const int*   pknn   = (const int*)  d_in[2];  // [B,N,NNB]
    const float* kpts   = (const float*)d_in[3];  // [KP,3]
    const float* wgt    = (const float*)d_in[4];  // [KP,C,C]
    const float* gamma  = (const float*)d_in[5];
    const float* beta   = (const float*)d_in[6];

    char* ws = (char*)d_ws;
    unsigned short* xt  = (unsigned short*)(ws + XT_OFF);
    unsigned short* wbT = (unsigned short*)(ws + WBT_OFF);
    float* scale = (float*)(ws + SCALE_OFF);
    float* psum  = (float*)(ws + PSUM_OFF);
    float* psq   = (float*)(ws + PSQ_OFF);
    float* tmp   = (float*)(ws + TMP_OFF);
    float* out   = (float*)d_out;

    k_xt<<<dim3(NPTS / 32, CC / 32, BATCH), dim3(32, 8), 0, stream>>>(x, xt);
    k_wt<<<(KP * CC * CC + 255) / 256, 256, 0, stream>>>(wgt, wbT, xt);
    kpconv_main<<<dim3(NPTS / MT, BATCH), 256, 0, stream>>>(pxyz, pknn, kpts, xt, wbT, tmp);
    k_stats<<<(BATCH * NPTS) / 64, 256, 0, stream>>>(tmp, psum, psq);
    k_finalize<<<CC, 256, 0, stream>>>(psum, psq, gamma, beta, scale);
    k_apply<<<dim3(NPTS / 32, CC / 32, BATCH), dim3(32, 8), 0, stream>>>(tmp, scale, out);
}
